// MSBM_81552839016900
// MI455X (gfx1250) — compile-verified, matrix operand order checked by reading, not run on hardware
//
#include <hip/hip_runtime.h>
#include <math.h>

typedef __attribute__((ext_vector_type(16))) _Float16 v16h;
typedef __attribute__((ext_vector_type(8)))  _Float16 v8h;
typedef __attribute__((ext_vector_type(8)))  float    v8f;
typedef long long i64;

#define LDT 40  // padded halfs per LDS tile row (80B, 16B-aligned chunks)

// ---------------- CDNA5 async copy: global -> LDS, 16B per lane -------------
static __device__ __forceinline__ void async_ld_b128(unsigned lds_off, const void* g) {
  asm volatile("global_load_async_to_lds_b128 %0, %1, off"
               :: "v"(lds_off), "v"(g) : "memory");
}
static __device__ __forceinline__ void wait_async0() {
  asm volatile("s_wait_asynccnt 0x0" ::: "memory");
}

// ---------------- WMMA fragment loader (16x32 f16, A/B symmetric) ------------
static __device__ __forceinline__ v16h frag_ld(const _Float16* base, int lane) {
  const int r  = lane & 15;
  const int k0 = (lane & 16) ? 8 : 0;
  const _Float16* q = base + r * LDT + k0;
  v8h lo = *(const v8h*)q;
  v8h hi = *(const v8h*)(q + 16);
  v16h out;
#pragma unroll
  for (int e = 0; e < 8; ++e) { out[e] = lo[e]; out[e + 8] = hi[e]; }
  return out;
}

// ------- Generic GEMM: C[M,N](f32) = A[M,K](f16) @ W[N,K]^T + bias ----------
// Double-buffered LDS staged with global_load_async_to_lds_b128 (ASYNCcnt).
__global__ void __launch_bounds__(256)
k_gemm_f16(const _Float16* __restrict__ A, const _Float16* __restrict__ W,
           const float* __restrict__ bias, float* __restrict__ C,
           int M, int N, int K) {
  __shared__ __align__(16) _Float16 As[2][128 * LDT];
  __shared__ __align__(16) _Float16 Bs[2][128 * LDT];
  const int tid = threadIdx.x, lane = tid & 31, wave = tid >> 5;
  const int wm = wave >> 2, wn = wave & 3;         // 2x4 wave grid: 64x32 per wave
  const i64 bm = (i64)blockIdx.y * 128;
  const int bn = blockIdx.x * 128;
  const int lr = tid >> 1, lc = (tid & 1) * 16;    // loader: row, half-chunk

  // Per-thread async-copy source pointers (16-half chunk each).
  // B rows >= N are clamped to N-1: output col j depends only on B row j, and
  // columns >= N are never stored, so clamped garbage is harmless.
  const _Float16* aptr = A + (bm + lr) * (i64)K + lc;
  const int nrow = (bn + lr < N) ? (bn + lr) : (N - 1);
  const _Float16* bptr = W + (i64)nrow * K + lc;
  const unsigned a0 = (unsigned)(unsigned long long)&As[0][0];
  const unsigned b0 = (unsigned)(unsigned long long)&Bs[0][0];
  const unsigned coff = (unsigned)(lr * LDT + lc) * 2u;  // byte offset in tile
  const unsigned bufb = (unsigned)(128 * LDT * 2);       // bytes per buffer

  const v8f zf = {};
  v8f acc[4][2];
#pragma unroll
  for (int i = 0; i < 4; ++i)
#pragma unroll
    for (int j = 0; j < 2; ++j) acc[i][j] = zf;

  // prologue: stage tile 0
  async_ld_b128(a0 + coff,      aptr);
  async_ld_b128(a0 + coff + 16, aptr + 8);
  async_ld_b128(b0 + coff,      bptr);
  async_ld_b128(b0 + coff + 16, bptr + 8);
  wait_async0();
  __syncthreads();

  int cur = 0;
  for (int k0 = 0; k0 < K; k0 += 32) {
    const int nxt = cur ^ 1;
    if (k0 + 32 < K) {  // overlap: async-stage next tile while computing
      const _Float16* an = aptr + k0 + 32;
      const _Float16* wn2 = bptr + k0 + 32;
      async_ld_b128(a0 + nxt * bufb + coff,      an);
      async_ld_b128(a0 + nxt * bufb + coff + 16, an + 8);
      async_ld_b128(b0 + nxt * bufb + coff,      wn2);
      async_ld_b128(b0 + nxt * bufb + coff + 16, wn2 + 8);
      if (k0 + 64 < K) __builtin_prefetch(bptr + k0 + 64, 0, 1);
    }
    v16h bf[2];
#pragma unroll
    for (int j = 0; j < 2; ++j)
      bf[j] = frag_ld(&Bs[cur][(wn * 32 + j * 16) * LDT], lane);
#pragma unroll
    for (int i = 0; i < 4; ++i) {
      v16h af = frag_ld(&As[cur][(wm * 64 + i * 16) * LDT], lane);
#pragma unroll
      for (int j = 0; j < 2; ++j)
        acc[i][j] = __builtin_amdgcn_wmma_f32_16x16x32_f16(
            false, af, false, bf[j], (short)0, acc[i][j], false, false);
    }
    wait_async0();      // our async copies for next tile done
    __syncthreads();    // all waves done reading cur + done staging nxt
    cur = nxt;
  }

  const int lrow = (lane >> 4) * 8, lcol = lane & 15;
#pragma unroll
  for (int i = 0; i < 4; ++i)
#pragma unroll
    for (int j = 0; j < 2; ++j) {
      const int col = bn + wn * 32 + j * 16 + lcol;
      if (col < N) {
        const float bv = bias ? bias[col] : 0.f;
        const i64 row0 = bm + wm * 64 + i * 16 + lrow;
#pragma unroll
        for (int r = 0; r < 8; ++r)
          C[(row0 + r) * (i64)N + col] = acc[i][j][r] + bv;
      }
    }
}

// -------- Strided temporal conv as virtual-im2col WMMA GEMM (N=256) ---------
// Synchronous staging kept here: out-of-range time rows need zero-fill.
__global__ void __launch_bounds__(256)
k_conv_gemm(const _Float16* __restrict__ X,   // (8, T, 1024) f16
            const _Float16* __restrict__ W,   // (256, taps*1024) f16, tap-major
            const float* __restrict__ bias, float* __restrict__ C,
            int Ls, int T, int Ktot, int stride, int pad) {
  __shared__ __align__(16) _Float16 As[128 * LDT];
  __shared__ __align__(16) _Float16 Bs[128 * LDT];
  const int tid = threadIdx.x, lane = tid & 31, wave = tid >> 5;
  const int wm = wave >> 2, wn = wave & 3;
  const i64 bm = (i64)blockIdx.y * 128;
  const int bn = blockIdx.x * 128;
  const int lr = tid >> 1, lc = (tid & 1) * 16;
  const int arow = (int)(bm + lr);
  const int b = arow / Ls, l = arow % Ls;

  const v8f zf = {};
  const v8h zh = {};
  v8f acc[4][2];
#pragma unroll
  for (int i = 0; i < 4; ++i)
#pragma unroll
    for (int j = 0; j < 2; ++j) acc[i][j] = zf;

  for (int k0 = 0; k0 < Ktot; k0 += 32) {
    {  // gathered A tile: K = (tap, channel)
      const int kg = k0 + lc;
      const int tap = kg >> 10;
      const int c = kg & 1023;
      const int tm = l * stride - pad + tap;
      v8h d0 = zh, d1 = zh;
      if (tm >= 0 && tm < T) {
        const _Float16* src = X + (((i64)b * T + tm) << 10) + c;
        d0 = *(const v8h*)src;
        d1 = *(const v8h*)(src + 8);
      }
      *(v8h*)&As[lr * LDT + lc]     = d0;
      *(v8h*)&As[lr * LDT + lc + 8] = d1;
    }
    {  // weight tile (N=256, always in range)
      const _Float16* src = W + (i64)(bn + lr) * Ktot + k0 + lc;
      *(v8h*)&Bs[lr * LDT + lc]     = *(const v8h*)(src);
      *(v8h*)&Bs[lr * LDT + lc + 8] = *(const v8h*)(src + 8);
      if (k0 + 32 < Ktot) __builtin_prefetch(src + 32, 0, 1);
    }
    __syncthreads();
    v16h bf[2];
#pragma unroll
    for (int j = 0; j < 2; ++j) bf[j] = frag_ld(&Bs[(wn * 32 + j * 16) * LDT], lane);
#pragma unroll
    for (int i = 0; i < 4; ++i) {
      v16h af = frag_ld(&As[(wm * 64 + i * 16) * LDT], lane);
#pragma unroll
      for (int j = 0; j < 2; ++j)
        acc[i][j] = __builtin_amdgcn_wmma_f32_16x16x32_f16(
            false, af, false, bf[j], (short)0, acc[i][j], false, false);
    }
    __syncthreads();
  }

  const int lrow = (lane >> 4) * 8, lcol = lane & 15;
#pragma unroll
  for (int i = 0; i < 4; ++i)
#pragma unroll
    for (int j = 0; j < 2; ++j) {
      const int col = bn + wn * 32 + j * 16 + lcol;
      const float bv = bias[col];
      const i64 row0 = bm + wm * 64 + i * 16 + lrow;
#pragma unroll
      for (int r = 0; r < 8; ++r)
        C[(row0 + r) * (i64)256 + col] = acc[i][j][r] + bv;
    }
}

// ---------------- SSD sequential scan; p split across wave32 blocks ---------
template <int NS>
__global__ void __launch_bounds__(32)
k_scan(const float* __restrict__ xbc, int xld,          // (b,L,convdim): x|B|C
       const float* __restrict__ dt,                    // (b,L,16)
       const float* __restrict__ A_log, const float* __restrict__ Dp,
       float* __restrict__ y,                           // (b,L,1024)
       int L, int rev) {
  const int pc = blockIdx.x & 1;
  const int h  = (blockIdx.x >> 1) & 15;
  const int b  = blockIdx.x >> 5;
  const int p  = pc * 32 + threadIdx.x;
  __shared__ float Bs[NS];
  __shared__ float Cs[NS];
  float S[NS];
#pragma unroll
  for (int n = 0; n < NS; ++n) S[n] = 0.f;
  const float Ah = -expf(A_log[h]);
  const float Dh = Dp[h];
  for (int step = 0; step < L; ++step) {
    const int t = rev ? (L - 1 - step) : step;
    const float* row = xbc + ((i64)b * L + t) * xld;
    for (int n = threadIdx.x; n < NS; n += 32) {
      Bs[n] = row[1024 + n];
      Cs[n] = row[1024 + NS + n];
    }
    const float xt  = row[h * 64 + p];
    const float dtv = dt[((i64)b * L + t) * 16 + h];
    const float dA  = expf(dtv * Ah);
    const float xdt = xt * dtv;
    __syncthreads();
    float accv = 0.f;
#pragma unroll
    for (int n = 0; n < NS; ++n) {
      const float sn = S[n] * dA + xdt * Bs[n];
      S[n] = sn;
      accv += sn * Cs[n];
    }
    y[((i64)b * L + t) * 1024 + h * 64 + p] = accv + xt * Dh;
    __syncthreads();
  }
}

// ---------------- Elementwise / norm kernels --------------------------------
__global__ void k_cast(const float* __restrict__ s, _Float16* __restrict__ d, i64 n) {
  const i64 i = (i64)blockIdx.x * blockDim.x + threadIdx.x;
  if (i < n) d[i] = (_Float16)s[i];
}

// (O=256,C=1024,K) -> f16 (O, K*1024) tap-major
__global__ void k_reorder_tw(const float* __restrict__ w, _Float16* __restrict__ d,
                             int K, i64 n) {
  const i64 i = (i64)blockIdx.x * blockDim.x + threadIdx.x;
  if (i >= n) return;
  const int kq = (int)(i % K);
  const i64 oc = i / K;
  const int c = (int)(oc & 1023);
  const int o = (int)(oc >> 10);
  d[(i64)o * (K * 1024) + kq * 1024 + c] = (_Float16)w[i];
}

__global__ void __launch_bounds__(256)
k_ln(const float* __restrict__ x, const float* __restrict__ w,
     const float* __restrict__ b, const float* __restrict__ res,
     float* __restrict__ y, _Float16* __restrict__ y16, int D) {
  const i64 row = blockIdx.x;
  const float* xr = x + row * (i64)D;
  __shared__ float sm[256];
  float s = 0.f;
  for (int i = threadIdx.x; i < D; i += 256) s += xr[i];
  sm[threadIdx.x] = s; __syncthreads();
  for (int o = 128; o > 0; o >>= 1) {
    if ((int)threadIdx.x < o) sm[threadIdx.x] += sm[threadIdx.x + o];
    __syncthreads();
  }
  const float mu = sm[0] / D;
  __syncthreads();
  float s2 = 0.f;
  for (int i = threadIdx.x; i < D; i += 256) { const float d0 = xr[i] - mu; s2 += d0 * d0; }
  sm[threadIdx.x] = s2; __syncthreads();
  for (int o = 128; o > 0; o >>= 1) {
    if ((int)threadIdx.x < o) sm[threadIdx.x] += sm[threadIdx.x + o];
    __syncthreads();
  }
  const float rstd = rsqrtf(sm[0] / D + 1e-5f);
  for (int i = threadIdx.x; i < D; i += 256) {
    float v = (xr[i] - mu) * rstd * w[i] + b[i];
    if (res) v += res[row * (i64)D + i];
    if (y)   y[row * (i64)D + i] = v;
    if (y16) y16[row * (i64)D + i] = (_Float16)v;
  }
}

__global__ void k_dt(const float* __restrict__ zx, int ld, int off,
                     const float* __restrict__ dtb, float* __restrict__ dt, i64 n) {
  const i64 i = (i64)blockIdx.x * blockDim.x + threadIdx.x;
  if (i >= n) return;
  const int h = (int)(i & 15);
  const i64 rowi = i >> 4;
  const float v = zx[rowi * (i64)ld + off + h] + dtb[h];
  dt[i] = (v > 20.f) ? v : log1pf(expf(v));
}

__global__ void k_dwconv(const float* __restrict__ zx, int ld, int off,
                         const float* __restrict__ w, const float* __restrict__ bias,
                         float* __restrict__ out, int L, int CD, int rev, i64 n) {
  const i64 i = (i64)blockIdx.x * blockDim.x + threadIdx.x;
  if (i >= n) return;
  const int c = (int)(i % CD);
  const i64 bl = i / CD;
  const int l = (int)(bl % L);
  const int b = (int)(bl / L);
  float a = bias[c];
#pragma unroll
  for (int j = 0; j < 4; ++j) {
    const int t = rev ? (l + 3 - j) : (l - 3 + j);
    if (t >= 0 && t < L) a += w[c * 4 + j] * zx[((i64)b * L + t) * ld + off + c];
  }
  out[i] = a / (1.f + expf(-a));  // silu
}

__global__ void __launch_bounds__(256)
k_gate_rms(const float* __restrict__ yin, const float* __restrict__ zx, int zld,
           const float* __restrict__ nw, _Float16* __restrict__ out) {
  const i64 row = blockIdx.x;
  const float* yr = yin + row * 1024;
  const float* zr = zx + row * (i64)zld;
  __shared__ float sm[256];
  float vals[4];
  float s2 = 0.f;
#pragma unroll
  for (int v = 0; v < 4; ++v) {
    const int i = threadIdx.x + v * 256;
    const float z = zr[i];
    const float g = z / (1.f + expf(-z));
    const float val = yr[i] * g;
    vals[v] = val;
    s2 += val * val;
  }
  sm[threadIdx.x] = s2; __syncthreads();
  for (int o = 128; o > 0; o >>= 1) {
    if ((int)threadIdx.x < o) sm[threadIdx.x] += sm[threadIdx.x + o];
    __syncthreads();
  }
  const float r = rsqrtf(sm[0] / 1024.f + 1e-5f);
#pragma unroll
  for (int v = 0; v < 4; ++v) {
    const int i = threadIdx.x + v * 256;
    out[row * 1024 + i] = (_Float16)(vals[v] * r * nw[i]);
  }
}

__global__ void k_combine(const float* __restrict__ a, const float* __restrict__ bq,
                          const float* __restrict__ x2, _Float16* __restrict__ out, i64 n) {
  const i64 i = (i64)blockIdx.x * blockDim.x + threadIdx.x;
  if (i >= n) return;
  const float z = x2[i];
  const float g = z / (1.f + expf(-z));
  out[i] = (_Float16)((a[i] + bq[i]) * g);
}

__global__ void k_interp(const float* __restrict__ enc, float* __restrict__ out,
                         int Ls, int first, i64 n) {
  const i64 i = (i64)blockIdx.x * blockDim.x + threadIdx.x;
  if (i >= n) return;
  const int d = (int)(i % 256);
  const i64 btl = i / 256;
  const int t = (int)(btl % 2048);
  const int b = (int)(btl / 2048);
  float pos = ((float)t + 0.5f) * ((float)Ls / 2048.f) - 0.5f;
  pos = fminf(fmaxf(pos, 0.f), (float)(Ls - 1));
  const int lo = (int)pos;
  const int hi = (lo + 1 < Ls) ? lo + 1 : Ls - 1;
  const float w = pos - (float)lo;
  const float v = enc[((i64)b * Ls + lo) * 256 + d] * (1.f - w) +
                  enc[((i64)b * Ls + hi) * 256 + d] * w;
  if (first) out[i] = v; else out[i] += v;
}

// ---------------- Host orchestration ----------------------------------------
static inline unsigned ucdiv(i64 a, i64 b) { return (unsigned)((a + b - 1) / b); }

extern "C" void kernel_launch(void* const* d_in, const int* in_sizes, int n_in,
                              void* d_out, int out_size, void* d_ws, size_t ws_size,
                              hipStream_t stream) {
  (void)in_sizes; (void)n_in; (void)out_size; (void)ws_size;
  const int Bt = 8, T = 2048;
  const float* x = (const float*)d_in[0];
  float* outp = (float*)d_out;

  char* ws = (char*)d_ws;
  size_t off = 0;
  auto alloc = [&](size_t bytes) -> void* {
    void* p = ws + off;
    off += (bytes + 255) & ~(size_t)255;
    return p;
  };

  _Float16* xf16 = (_Float16*)alloc((size_t)Bt * T * 1024 * 2);
  {
    const i64 n = (i64)Bt * T * 1024;
    k_cast<<<ucdiv(n, 256), 256, 0, stream>>>(x, xf16, n);
  }
  const size_t scale_base = off;

  for (int si = 0; si < 2; ++si) {
    off = scale_base;
    const int s = (si == 0) ? 2 : 4;
    const int kk = s + 1;                       // s even -> k = s+1 (3, 5)
    const int pad = (kk - s + 1) / 2;           // 1, 1
    const int Ls = (T + 2 * pad - kk) / s + 1;  // 1024, 512
    const int ds = 128 + 16 * s;                // 160, 192
    const int dinproj = 2048 + 2 * ds + 16;     // 2384, 2448
    const int convdim = 1024 + 2 * ds;          // 1344, 1408
    const int M = Bt * Ls;                      // 8192, 4096

    void* const* P = d_in + 1 + si * 28;
    const float* t_convw  = (const float*)P[0];
    const float* t_convb  = (const float*)P[1];
    const float* t_lnw    = (const float*)P[2];
    const float* t_lnb    = (const float*)P[3];
    const float* pre_w    = (const float*)P[4];
    const float* pre_b    = (const float*)P[5];
    const float* pre_lnw  = (const float*)P[6];
    const float* pre_lnb  = (const float*)P[7];
    const float* post_w   = (const float*)P[8];
    const float* post_b   = (const float*)P[9];
    const float* post_lnw = (const float*)P[10];
    const float* post_lnb = (const float*)P[11];

    _Float16* wt16    = (_Float16*)alloc((size_t)256 * kk * 1024 * 2);
    float*    featraw = (float*)alloc((size_t)M * 256 * 4);
    float*    feat    = (float*)alloc((size_t)M * 256 * 4);
    _Float16* feat16  = (_Float16*)alloc((size_t)M * 256 * 2);
    _Float16* prew16  = (_Float16*)alloc((size_t)512 * 256 * 2);
    float*    preraw  = (float*)alloc((size_t)M * 512 * 4);
    float*    x2      = (float*)alloc((size_t)M * 512 * 4);
    _Float16* x216    = (_Float16*)alloc((size_t)M * 512 * 2);
    _Float16* wip16   = (_Float16*)alloc((size_t)dinproj * 512 * 2);
    float*    zx      = (float*)alloc((size_t)M * dinproj * 4);
    float*    dtb     = (float*)alloc((size_t)M * 16 * 4);
    float*    xbc     = (float*)alloc((size_t)M * convdim * 4);
    float*    ybuf    = (float*)alloc((size_t)M * 1024 * 4);
    _Float16* y16     = (_Float16*)alloc((size_t)M * 1024 * 2);
    _Float16* wop16   = (_Float16*)alloc((size_t)512 * 1024 * 2);
    float*    outd0   = (float*)alloc((size_t)M * 512 * 4);
    float*    outd1   = (float*)alloc((size_t)M * 512 * 4);
    _Float16* ycmb16  = (_Float16*)alloc((size_t)M * 512 * 2);
    _Float16* wpost16 = (_Float16*)alloc((size_t)256 * 512 * 2);
    float*    postraw = featraw;  // reuse (featraw dead after LN)
    float*    enc     = preraw;   // reuse (preraw dead after LN)
    float*    outd[2] = {outd0, outd1};

    // temporal conv (im2col WMMA GEMM) + LN
    {
      const i64 n = (i64)256 * 1024 * kk;
      k_reorder_tw<<<ucdiv(n, 256), 256, 0, stream>>>(t_convw, wt16, kk, n);
    }
    k_conv_gemm<<<dim3(2, M / 128), 256, 0, stream>>>(xf16, wt16, t_convb, featraw,
                                                      Ls, T, kk * 1024, s, pad);
    k_ln<<<M, 256, 0, stream>>>(featraw, t_lnw, t_lnb, nullptr, feat, feat16, 256);

    // pre projection + LN (x2 is both mamba input and SiLU gate source)
    k_cast<<<ucdiv((i64)512 * 256, 256), 256, 0, stream>>>(pre_w, prew16, (i64)512 * 256);
    k_gemm_f16<<<dim3(4, M / 128), 256, 0, stream>>>(feat16, prew16, pre_b, preraw,
                                                     M, 512, 256);
    k_ln<<<M, 256, 0, stream>>>(preraw, pre_lnw, pre_lnb, nullptr, x2, x216, 512);

    // bidirectional Mamba2 (dir=1 runs time-reversed via rev flags; no flips)
    for (int dir = 0; dir < 2; ++dir) {
      const float* m_ip  = (const float*)P[12 + dir * 8 + 0];
      const float* m_cw  = (const float*)P[12 + dir * 8 + 1];
      const float* m_cb  = (const float*)P[12 + dir * 8 + 2];
      const float* m_dtb = (const float*)P[12 + dir * 8 + 3];
      const float* m_al  = (const float*)P[12 + dir * 8 + 4];
      const float* m_D   = (const float*)P[12 + dir * 8 + 5];
      const float* m_nw  = (const float*)P[12 + dir * 8 + 6];
      const float* m_op  = (const float*)P[12 + dir * 8 + 7];

      k_cast<<<ucdiv((i64)dinproj * 512, 256), 256, 0, stream>>>(m_ip, wip16,
                                                                 (i64)dinproj * 512);
      k_gemm_f16<<<dim3(ucdiv(dinproj, 128), M / 128), 256, 0, stream>>>(
          x216, wip16, nullptr, zx, M, dinproj, 512);
      k_dt<<<ucdiv((i64)M * 16, 256), 256, 0, stream>>>(zx, dinproj, 2048 + 2 * ds,
                                                        m_dtb, dtb, (i64)M * 16);
      k_dwconv<<<ucdiv((i64)M * convdim, 256), 256, 0, stream>>>(
          zx, dinproj, 1024, m_cw, m_cb, xbc, Ls, convdim, dir, (i64)M * convdim);
      if (ds == 160)
        k_scan<160><<<dim3(Bt * 16 * 2), 32, 0, stream>>>(xbc, convdim, dtb, m_al,
                                                          m_D, ybuf, Ls, dir);
      else
        k_scan<192><<<dim3(Bt * 16 * 2), 32, 0, stream>>>(xbc, convdim, dtb, m_al,
                                                          m_D, ybuf, Ls, dir);
      k_gate_rms<<<M, 256, 0, stream>>>(ybuf, zx, dinproj, m_nw, y16);
      k_cast<<<ucdiv((i64)512 * 1024, 256), 256, 0, stream>>>(m_op, wop16,
                                                              (i64)512 * 1024);
      k_gemm_f16<<<dim3(4, M / 128), 256, 0, stream>>>(y16, wop16, nullptr,
                                                       outd[dir], M, 512, 1024);
    }

    // combine with gate, post projection + LN + residual, interpolate-add
    k_combine<<<ucdiv((i64)M * 512, 256), 256, 0, stream>>>(outd[0], outd[1], x2,
                                                            ycmb16, (i64)M * 512);
    k_cast<<<ucdiv((i64)256 * 512, 256), 256, 0, stream>>>(post_w, wpost16,
                                                           (i64)256 * 512);
    k_gemm_f16<<<dim3(2, M / 128), 256, 0, stream>>>(ycmb16, wpost16, post_b,
                                                     postraw, M, 256, 512);
    k_ln<<<M, 256, 0, stream>>>(postraw, post_lnw, post_lnb, feat, enc, nullptr, 256);
    k_interp<<<ucdiv((i64)Bt * T * 256, 256), 256, 0, stream>>>(
        enc, outp, Ls, si == 0 ? 1 : 0, (i64)Bt * T * 256);
  }
}
